// QueryTransformer_46033459478663
// MI455X (gfx1250) — matrix operand order repaired and checked by reading, not compiled, not run on hardware
//
#include <hip/hip_runtime.h>
#include <hip/hip_fp16.h>

typedef _Float16 f16;
typedef __attribute__((ext_vector_type(16))) _Float16 v16h;
typedef __attribute__((ext_vector_type(8)))  _Float16 v8h;
typedef __attribute__((ext_vector_type(8)))  float    v8f;

// -------------------- elementwise helpers --------------------

__device__ __forceinline__ float phi_elu1(float x) {   // elu(x)+1
    return x > 0.f ? x + 1.f : __expf(x);
}

// -------------------- [B,C,L] -> [B,L,C] transpose --------------------
__global__ __launch_bounds__(256) void transpose_kernel(
    const float* __restrict__ in, float* __restrict__ out, int C, int L)
{
    __shared__ float tile[32][33];
    int b  = blockIdx.z;
    int l0 = blockIdx.x * 32, c0 = blockIdx.y * 32;
    #pragma unroll
    for (int i = 0; i < 4; i++) {
        int c = c0 + threadIdx.y + i * 8;
        int l = l0 + threadIdx.x;
        tile[threadIdx.y + i * 8][threadIdx.x] = in[((size_t)b * C + c) * L + l];
    }
    __syncthreads();
    #pragma unroll
    for (int i = 0; i < 4; i++) {
        int c = c0 + threadIdx.x;
        int l = l0 + threadIdx.y + i * 8;
        out[((size_t)b * L + l) * C + c] = tile[threadIdx.x][threadIdx.y + i * 8];
    }
}

// -------------------- fused LayerNorm (+pos add) -> f16 --------------------
// one wave per row of 256; outA = f16(LN(x)+pos), outB (optional) = f16(LN(x))
__global__ __launch_bounds__(256) void ln_kernel(
    const float* __restrict__ x, const float* __restrict__ g, const float* __restrict__ bb,
    const float* __restrict__ pos, int posM,
    f16* __restrict__ outA, f16* __restrict__ outB, int M)
{
    int wid = threadIdx.x >> 5, lane = threadIdx.x & 31;
    int row = blockIdx.x * 8 + wid;
    if (row >= M) return;
    const float* xr = x + (size_t)row * 256 + lane * 8;
    float4 q0 = *(const float4*)xr;
    float4 q1 = *(const float4*)(xr + 4);
    float v[8] = {q0.x, q0.y, q0.z, q0.w, q1.x, q1.y, q1.z, q1.w};
    float s = 0.f, sq = 0.f;
    #pragma unroll
    for (int j = 0; j < 8; j++) { s += v[j]; sq += v[j] * v[j]; }
    #pragma unroll
    for (int off = 16; off > 0; off >>= 1) {
        s  += __shfl_xor(s,  off, 32);
        sq += __shfl_xor(sq, off, 32);
    }
    float mean = s * (1.f / 256.f);
    float var  = sq * (1.f / 256.f) - mean * mean;
    float rstd = rsqrtf(var + 1e-5f);
    int prow = row % posM;
    #pragma unroll
    for (int j = 0; j < 8; j++) {
        int c = lane * 8 + j;
        float y = (v[j] - mean) * rstd * g[c] + bb[c];
        if (outB) outB[(size_t)row * 256 + c] = (f16)y;
        float ya = y + (pos ? pos[(size_t)prow * 256 + c] : 0.f);
        outA[(size_t)row * 256 + c] = (f16)ya;
    }
}

// -------------------- weight transpose + cast: wt[n*K+k] = w[k*N+n] --------------------
__global__ __launch_bounds__(256) void wcast_kernel(
    const float* __restrict__ w, f16* __restrict__ wt, int K, int N)
{
    int i = blockIdx.x * 256 + threadIdx.x;
    if (i >= K * N) return;
    int n = i / K, k = i - n * K;
    wt[i] = (f16)w[(size_t)k * N + n];
}

// -------------------- elementwise (a [+ b]) -> f16 --------------------
__global__ __launch_bounds__(256) void addcast_kernel(
    const float* __restrict__ a, const float* __restrict__ b, f16* __restrict__ o, int n)
{
    int i = blockIdx.x * 256 + threadIdx.x;
    if (i >= n) return;
    float v = a[i];
    if (b) v += b[i];
    o[i] = (f16)v;
}

// -------------------- WMMA GEMM: out = act(A @ Bt^T + bias) + res --------------------
// A: [M,K] f16 row-major.  Bt: [N,K] f16 (weights pre-transposed to n-major).
// 256 threads = 8 waves; block tile 64x64; each wave: two 16x16 accumulators.
// act: 0=none, 1=gelu(exact), 2=relu.  Writes fp32 and/or f16.
__global__ __launch_bounds__(256) void gemm_wmma_kernel(
    const f16* __restrict__ A, const f16* __restrict__ Bt,
    const float* __restrict__ bias, const float* __restrict__ res,
    float* __restrict__ out32, f16* __restrict__ out16,
    int M, int N, int K, int act)
{
    int tid  = threadIdx.x;
    int wid  = tid >> 5, lane = tid & 31;
    int m0   = blockIdx.x * 64 + (wid & 3) * 16;
    int n0   = blockIdx.y * 64 + (wid >> 2) * 32;
    int half = lane >> 4, l16 = lane & 15;

    int mA = m0 + l16; if (mA >= M) mA = M - 1;   // clamp (rows guarded at store)
    int nB0 = n0 + l16;
    int nB1 = nB0 + 16;

    // per-lane fragment base pointers (all 16B-aligned contiguous chunks)
    const f16* arow  = A  + (size_t)mA  * K + half * 16;   // A frag: k = kk+half*16+{0..15}
    const f16* brow0 = Bt + (size_t)nB0 * K + half * 16;   // B frag: k = kk+half*16+{0..15}
    const f16* brow1 = Bt + (size_t)nB1 * K + half * 16;

    v8f c0 = {}; v8f c1 = {};
    for (int kk = 0; kk < K; kk += 32) {
        v8h alo  = *(const v8h*)(arow  + kk);
        v8h ahi  = *(const v8h*)(arow  + kk + 8);
        v8h b0lo = *(const v8h*)(brow0 + kk);
        v8h b0hi = *(const v8h*)(brow0 + kk + 8);
        v8h b1lo = *(const v8h*)(brow1 + kk);
        v8h b1hi = *(const v8h*)(brow1 + kk + 8);
        v16h av, bv0, bv1;
        #pragma unroll
        for (int j = 0; j < 8; j++) {
            av[j]  = alo[j];  av[j + 8]  = ahi[j];
            bv0[j] = b0lo[j]; bv0[j + 8] = b0hi[j];
            bv1[j] = b1lo[j]; bv1[j + 8] = b1hi[j];
        }
        c0 = __builtin_amdgcn_wmma_f32_16x16x32_f16(false, av, false, bv0, (short)0, c0, false, false);
        c1 = __builtin_amdgcn_wmma_f32_16x16x32_f16(false, av, false, bv1, (short)0, c1, false, false);
    }

    int rowbase = m0 + half * 8;
    float bias0 = bias ? bias[nB0] : 0.f;
    float bias1 = bias ? bias[nB1] : 0.f;
    #pragma unroll
    for (int r = 0; r < 8; r++) {
        int row = rowbase + r;
        if (row >= M) continue;
        float v0 = c0[r] + bias0;
        float v1 = c1[r] + bias1;
        if (act == 1) {
            v0 = 0.5f * v0 * (1.f + erff(v0 * 0.70710678118f));
            v1 = 0.5f * v1 * (1.f + erff(v1 * 0.70710678118f));
        } else if (act == 2) {
            v0 = fmaxf(v0, 0.f);
            v1 = fmaxf(v1, 0.f);
        }
        if (res) {
            v0 += res[(size_t)row * N + nB0];
            v1 += res[(size_t)row * N + nB1];
        }
        if (out32) {
            out32[(size_t)row * N + nB0] = v0;
            out32[(size_t)row * N + nB1] = v1;
        }
        if (out16) {
            out16[(size_t)row * N + nB0] = (f16)v0;
            out16[(size_t)row * N + nB1] = (f16)v1;
        }
    }
}

// -------------------- linear-attention KV accumulation (partials) --------------------
// Kmat/Vmat: [B*Lk, 256] fp32 (head h at cols h*64..h*64+63).
// grid.x = 32*chunks (32 = B*H). Each block accumulates a chunk of Lk rows into
// KVp[chunk][bh][64][64] and KSp[chunk][bh][64].  KV uses V/Lk; Ksum = sum phi(K).
__global__ __launch_bounds__(256) void kv_accum_kernel(
    const float* __restrict__ Kmat, const float* __restrict__ Vmat,
    float* __restrict__ KVp, float* __restrict__ KSp, int Lk, int chunks)
{
    int bh = blockIdx.x / chunks;
    int ci = blockIdx.x % chunks;
    int b = bh >> 3, h = bh & 7;
    int clen = (Lk + chunks - 1) / chunks;
    int s0 = ci * clen;
    int s1 = min(Lk, s0 + clen);

    __shared__ float sK[8 * 64];
    __shared__ float sV[8 * 64];
    int t  = threadIdx.x;
    int d  = t >> 2;
    int vb = (t & 3) * 16;
    float acc[16];
    #pragma unroll
    for (int j = 0; j < 16; j++) acc[j] = 0.f;
    float ks = 0.f;

    for (int s = s0; s < s1; s += 8) {
        for (int i = t; i < 512; i += 256) {
            int r = i >> 6, c = i & 63;
            int srow = s + r;
            float kp = 0.f, vv = 0.f;
            if (srow < s1) {
                size_t base = (size_t)(b * Lk + srow) * 256 + h * 64 + c;
                kp = phi_elu1(Kmat[base]);
                vv = Vmat[base];
            }
            sK[i] = kp;
            sV[i] = vv;
        }
        __syncthreads();
        #pragma unroll
        for (int r = 0; r < 8; r++) {
            float kp = sK[r * 64 + d];
            ks += kp;
            const float* vr = &sV[r * 64 + vb];
            #pragma unroll
            for (int j = 0; j < 16; j++) acc[j] += kp * vr[j];
        }
        __syncthreads();
    }
    float inv = 1.f / (float)Lk;
    float* dst = KVp + ((size_t)ci * 32 + bh) * 4096 + d * 64 + vb;
    #pragma unroll
    for (int j = 0; j < 16; j++) dst[j] = acc[j] * inv;
    if ((t & 3) == 0) KSp[((size_t)ci * 32 + bh) * 64 + d] = ks;
}

// fixed-order partial reduction (deterministic, no atomics)
__global__ __launch_bounds__(256) void kv_reduce_kernel(
    const float* __restrict__ KVp, const float* __restrict__ KSp,
    float* __restrict__ KV, float* __restrict__ Ksum, int chunks)
{
    int i = blockIdx.x * 256 + threadIdx.x;
    if (i < 32 * 64 * 64) {
        float s = 0.f;
        for (int c = 0; c < chunks; c++) s += KVp[(size_t)c * 131072 + i];
        KV[i] = s;
    }
    if (i < 32 * 64) {
        float s = 0.f;
        for (int c = 0; c < chunks; c++) s += KSp[(size_t)c * 2048 + i];
        Ksum[i] = s;
    }
}

// -------------------- attention output: msg = phi(Q).KV * Z * Lk -> f16 --------------------
// grid.x = B*Lq (one block per query row). out16: [B*Lq, 256] f16.
__global__ __launch_bounds__(256) void attn_out_kernel(
    const float* __restrict__ Qmat, const float* __restrict__ KV,
    const float* __restrict__ Ksum, f16* __restrict__ out16, int Lq, int Lk)
{
    int row = blockIdx.x;
    int b = row / Lq;
    __shared__ float qphi[256];
    __shared__ float ks[512];
    int t = threadIdx.x;
    qphi[t] = phi_elu1(Qmat[(size_t)row * 256 + t]);
    ks[t]       = Ksum[(size_t)b * 512 + t];
    ks[t + 256] = Ksum[(size_t)b * 512 + 256 + t];
    __syncthreads();

    int h  = t >> 5;
    int v0 = (t & 31) * 2;
    float dot = 0.f;
    #pragma unroll
    for (int d = 0; d < 64; d++) dot += qphi[h * 64 + d] * ks[h * 64 + d];
    float Z = 1.f / (dot + 1e-6f);

    const float* kvb = KV + (size_t)(b * 8 + h) * 4096;
    float a0 = 0.f, a1 = 0.f;
    #pragma unroll
    for (int d = 0; d < 64; d++) {
        float q = qphi[h * 64 + d];
        a0 += q * kvb[d * 64 + v0];
        a1 += q * kvb[d * 64 + v0 + 1];
    }
    float scale = Z * (float)Lk;
    out16[(size_t)row * 256 + h * 64 + v0]     = (f16)(a0 * scale);
    out16[(size_t)row * 256 + h * 64 + v0 + 1] = (f16)(a1 * scale);
}

// ==================== host orchestration ====================

extern "C" void kernel_launch(void* const* d_in, const int* in_sizes, int n_in,
                              void* d_out, int out_size, void* d_ws, size_t ws_size,
                              hipStream_t stream) {
    (void)in_sizes; (void)n_in; (void)out_size; (void)ws_size;
    const int B_ = 4, L_ = 4096, NQ_ = 100;
    const int Mbig = B_ * L_;   // 16384
    const int Mq   = B_ * NQ_;  // 400

    // ---- inputs (jax pytree / alphabetical dict order) ----
    const float* feat0 = (const float*)d_in[0];
    const float* feat1 = (const float*)d_in[1];
    const float* qe0   = (const float*)d_in[2];
    const float* qe1   = (const float*)d_in[3];
    const float* pos0  = (const float*)d_in[4];
    const float* pos1  = (const float*)d_in[5];
    // dec params (base 6): ca_kb,ca_kw,ca_mw,ca_qb,ca_qw,ca_vb,ca_vw,m1,m2,
    //                      n1_b,n1_g,n2_b,n2_g,n3_b,n3_g,sa_kb,sa_kw,sa_mw,sa_qb,sa_qw,sa_vb,sa_vw
    const float* d_ca_kb = (const float*)d_in[6];
    const float* d_ca_kw = (const float*)d_in[7];
    const float* d_ca_mw = (const float*)d_in[8];
    const float* d_ca_qb = (const float*)d_in[9];
    const float* d_ca_qw = (const float*)d_in[10];
    const float* d_ca_vb = (const float*)d_in[11];
    const float* d_ca_vw = (const float*)d_in[12];
    const float* d_m1    = (const float*)d_in[13];
    const float* d_m2    = (const float*)d_in[14];
    const float* d_n1b   = (const float*)d_in[15];
    const float* d_n1g   = (const float*)d_in[16];
    const float* d_n2b   = (const float*)d_in[17];
    const float* d_n2g   = (const float*)d_in[18];
    const float* d_n3b   = (const float*)d_in[19];
    const float* d_n3g   = (const float*)d_in[20];
    const float* d_sa_kb = (const float*)d_in[21];
    const float* d_sa_kw = (const float*)d_in[22];
    const float* d_sa_mw = (const float*)d_in[23];
    const float* d_sa_qb = (const float*)d_in[24];
    const float* d_sa_qw = (const float*)d_in[25];
    const float* d_sa_vb = (const float*)d_in[26];
    const float* d_sa_vw = (const float*)d_in[27];
    // enc params (base 28): kw,m1,m2,mw,n2_b,n2_g,nkv_b,nkv_g,nq_b,nq_g,qw,vw
    const float* e_kw   = (const float*)d_in[28];
    const float* e_m1   = (const float*)d_in[29];
    const float* e_m2   = (const float*)d_in[30];
    const float* e_mw   = (const float*)d_in[31];
    const float* e_n2b  = (const float*)d_in[32];
    const float* e_n2g  = (const float*)d_in[33];
    const float* e_nkvb = (const float*)d_in[34];
    const float* e_nkvg = (const float*)d_in[35];
    const float* e_nqb  = (const float*)d_in[36];
    const float* e_nqg  = (const float*)d_in[37];
    const float* e_qw   = (const float*)d_in[38];
    const float* e_vw   = (const float*)d_in[39];

    // ---- workspace layout ----
    char* ws = (char*)d_ws;
    constexpr size_t SZ = (size_t)4 * 4096 * 256 * 4;  // 16 MB per [B,L,C] fp32
    float* f0  = (float*)(ws + 0 * SZ);
    float* f1  = (float*)(ws + 1 * SZ);
    float* p0  = (float*)(ws + 2 * SZ);
    float* p1  = (float*)(ws + 3 * SZ);
    float* fc0 = (float*)(ws + 4 * SZ);
    float* fc1 = (float*)(ws + 5 * SZ);
    float* Qb  = (float*)(ws + 6 * SZ);
    float* Kb  = (float*)(ws + 7 * SZ);
    float* Vb  = (float*)(ws + 8 * SZ);
    f16*   A16a = (f16*)(ws + 9 * SZ);            // 8 MB
    f16*   A16b = (f16*)(ws + 9 * SZ + SZ / 2);   // 8 MB
    f16*   H16  = (f16*)(ws + 10 * SZ);           // 16 MB  [B*L, 512] f16
    char*  b11  = ws + 11 * SZ;
    float* KVp  = (float*)(b11);                  // 4 MB: 8 chunks * 32 * 4096
    float* KSp  = (float*)(b11 + 0x400000);       // 64 KB
    float* KV   = (float*)(b11 + 0x410000);       // 512 KB
    float* Ksum = (float*)(b11 + 0x490000);       // 8 KB
    f16*   W16  = (f16*)(b11 + 0x4A0000);         // 512 KB (transposed f16 weights)
    float* h0   = (float*)(b11 + 0x520000);
    float* h1   = (float*)(b11 + 0x5A0000);
    f16*   qk16d  = (f16*)(b11 + 0x620000);
    f16*   t216d  = (f16*)(b11 + 0x6A0000);
    f16*   msg16d = (f16*)(b11 + 0x720000);
    f16*   Hd16   = (f16*)(b11 + 0x7A0000);

    // ---- launch helpers ----
    auto LN = [&](const float* x, const float* g, const float* bb, const float* pos,
                  int posM, f16* oA, f16* oB, int M) {
        ln_kernel<<<(M + 7) / 8, 256, 0, stream>>>(x, g, bb, pos, posM, oA, oB, M);
    };
    auto GEMM = [&](const f16* A, const float* bias, const float* res,
                    float* o32, f16* o16, int M, int N, int K, int act) {
        dim3 g((M + 63) / 64, N / 64);
        gemm_wmma_kernel<<<g, 256, 0, stream>>>(A, W16, bias, res, o32, o16, M, N, K, act);
    };
    auto WT = [&](const float* w, int K, int N) {
        int n = K * N;
        wcast_kernel<<<(n + 255) / 256, 256, 0, stream>>>(w, W16, K, N);
    };
    auto ATT = [&](f16* o16, int Lq, int Lk, int chunks) {
        kv_accum_kernel<<<32 * chunks, 256, 0, stream>>>(Kb, Vb, KVp, KSp, Lk, chunks);
        kv_reduce_kernel<<<512, 256, 0, stream>>>(KVp, KSp, KV, Ksum, chunks);
        attn_out_kernel<<<B_ * Lq, 256, 0, stream>>>(Qb, KV, Ksum, o16, Lq, Lk);
    };
    auto ADDCAST = [&](const float* a, const float* b, f16* o, int n) {
        addcast_kernel<<<(n + 255) / 256, 256, 0, stream>>>(a, b, o, n);
    };

    // ---- stage 0: [B,C,H*W] -> [B,L,C] ----
    {
        dim3 g(L_ / 32, 256 / 32, B_), blk(32, 8);
        transpose_kernel<<<g, blk, 0, stream>>>(feat0, f0, 256, L_);
        transpose_kernel<<<g, blk, 0, stream>>>(feat1, f1, 256, L_);
        transpose_kernel<<<g, blk, 0, stream>>>(pos0, p0, 256, L_);
        transpose_kernel<<<g, blk, 0, stream>>>(pos1, p1, 256, L_);
    }

    // ---- encoder: 8 layers, alternating self/cross ----
    for (int i = 0; i < 8; i++) {
        const size_t oW = (size_t)i * 65536, oF = (size_t)i * 131072, oN = (size_t)i * 256;
        bool cross = (i & 1);
        if (cross) {
            hipMemcpyAsync(fc0, f0, SZ, hipMemcpyDeviceToDevice, stream);
            hipMemcpyAsync(fc1, f1, SZ, hipMemcpyDeviceToDevice, stream);
        }
        for (int img = 0; img < 2; img++) {
            float* x = img ? f1 : f0;
            const float* xpos = img ? p1 : p0;
            const float* src, *spos;
            if (cross) { src = img ? fc0 : fc1; spos = img ? p0 : p1; }
            else       { src = x;              spos = xpos; }

            LN(x,   e_nqg + oN,  e_nqb + oN,  xpos, Mbig, A16a, nullptr, Mbig);
            LN(src, e_nkvg + oN, e_nkvb + oN, spos, Mbig, A16b, nullptr, Mbig);
            WT(e_qw + oW, 256, 256); GEMM(A16a, nullptr, nullptr, Qb, nullptr, Mbig, 256, 256, 0);
            WT(e_kw + oW, 256, 256); GEMM(A16b, nullptr, nullptr, Kb, nullptr, Mbig, 256, 256, 0);
            WT(e_vw + oW, 256, 256); GEMM(A16b, nullptr, nullptr, Vb, nullptr, Mbig, 256, 256, 0);
            ATT(A16a, L_, L_, 8);                                     // msg -> A16a (f16)
            WT(e_mw + oW, 256, 256); GEMM(A16a, nullptr, x, x, nullptr, Mbig, 256, 256, 0);
            LN(x, e_n2g + oN, e_n2b + oN, nullptr, Mbig, A16a, nullptr, Mbig);
            WT(e_m1 + oF, 256, 512); GEMM(A16a, nullptr, nullptr, nullptr, H16, Mbig, 512, 256, 1);
            WT(e_m2 + oF, 512, 256); GEMM(H16,  nullptr, x, x, nullptr, Mbig, 256, 512, 0);
        }
    }

    // ---- decoder: 2 layers ----
    hipMemsetAsync(h0, 0, (size_t)Mq * 256 * 4, stream);
    hipMemsetAsync(h1, 0, (size_t)Mq * 256 * 4, stream);
    for (int i = 0; i < 2; i++) {
        const size_t oW = (size_t)i * 65536, oF = (size_t)i * 131072, oN = (size_t)i * 256;
        for (int img = 0; img < 2; img++) {
            float* h = img ? h1 : h0;
            const float* fm = img ? f1 : f0;
            const float* qe = img ? qe1 : qe0;
            const float* pp = img ? p1 : p0;

            // self-attention (v input = tgt2 without pos)
            LN(h, d_n1g + oN, d_n1b + oN, qe, NQ_, qk16d, t216d, Mq);
            WT(d_sa_qw + oW, 256, 256); GEMM(qk16d, d_sa_qb + oN, nullptr, Qb, nullptr, Mq, 256, 256, 0);
            WT(d_sa_kw + oW, 256, 256); GEMM(qk16d, d_sa_kb + oN, nullptr, Kb, nullptr, Mq, 256, 256, 0);
            WT(d_sa_vw + oW, 256, 256); GEMM(t216d, d_sa_vb + oN, nullptr, Vb, nullptr, Mq, 256, 256, 0);
            ATT(msg16d, NQ_, NQ_, 1);
            WT(d_sa_mw + oW, 256, 256); GEMM(msg16d, nullptr, h, h, nullptr, Mq, 256, 256, 0);

            // cross-attention (k = memory+pos, v = memory)
            LN(h, d_n2g + oN, d_n2b + oN, qe, NQ_, qk16d, nullptr, Mq);
            ADDCAST(fm, pp, A16a, Mbig * 256);
            ADDCAST(fm, nullptr, A16b, Mbig * 256);
            WT(d_ca_qw + oW, 256, 256); GEMM(qk16d, d_ca_qb + oN, nullptr, Qb, nullptr, Mq,   256, 256, 0);
            WT(d_ca_kw + oW, 256, 256); GEMM(A16a,  d_ca_kb + oN, nullptr, Kb, nullptr, Mbig, 256, 256, 0);
            WT(d_ca_vw + oW, 256, 256); GEMM(A16b,  d_ca_vb + oN, nullptr, Vb, nullptr, Mbig, 256, 256, 0);
            ATT(msg16d, NQ_, L_, 8);
            WT(d_ca_mw + oW, 256, 256); GEMM(msg16d, nullptr, h, h, nullptr, Mq, 256, 256, 0);

            // FFN (relu)
            LN(h, d_n3g + oN, d_n3b + oN, nullptr, 1, qk16d, nullptr, Mq);
            WT(d_m1 + oF, 256, 512); GEMM(qk16d, nullptr, nullptr, nullptr, Hd16, Mq, 512, 256, 2);
            WT(d_m2 + oF, 512, 256); GEMM(Hd16,  nullptr, h, h, nullptr, Mq, 256, 512, 0);
        }
    }

    // ---- pack outputs: (h0, h1, f0, f1) ----
    float* out = (float*)d_out;
    const size_t hN = (size_t)Mq * 256;       // 102400
    const size_t fN = (size_t)Mbig * 256;     // 4194304
    hipMemcpyAsync(out,               h0, hN * 4, hipMemcpyDeviceToDevice, stream);
    hipMemcpyAsync(out + hN,          h1, hN * 4, hipMemcpyDeviceToDevice, stream);
    hipMemcpyAsync(out + 2 * hN,      f0, fN * 4, hipMemcpyDeviceToDevice, stream);
    hipMemcpyAsync(out + 2 * hN + fN, f1, fN * 4, hipMemcpyDeviceToDevice, stream);
}
